// global_aggregation_1211180777530
// MI455X (gfx1250) — matrix operand order repaired and checked by reading, not compiled, not run on hardware
//
#include <hip/hip_runtime.h>
#include <hip/hip_bf16.h>

#define F 256
#define G 8192
#define NEG_SLOPE 0.01f

typedef __attribute__((ext_vector_type(16))) __bf16 v16bf;
typedef __attribute__((ext_vector_type(8)))  float  v8f;

union ABfrag {
    v16bf v;
    unsigned u[8];
    uint4 q[2];
};

// ---- helpers -------------------------------------------------------------

// float -> bf16 (round to nearest even), returned in low 16 bits
__device__ __forceinline__ unsigned f2bf1(float f) {
    unsigned u = __float_as_uint(f);
    return (u + 0x7FFFu + ((u >> 16) & 1u)) >> 16;
}
__device__ __forceinline__ unsigned packbf(float lo, float hi) {
    return f2bf1(lo) | (f2bf1(hi) << 16);
}
// monotonic float->uint key (key 0 is below key(-inf): perfect "empty" sentinel)
__device__ __forceinline__ unsigned fkey(float f) {
    unsigned u = __float_as_uint(f);
    return (u & 0x80000000u) ? ~u : (u | 0x80000000u);
}
__device__ __forceinline__ float fdekey(unsigned k) {
    unsigned u = (k & 0x80000000u) ? (k ^ 0x80000000u) : ~k;
    return __uint_as_float(u);
}

// async global -> LDS copy, 16 bytes per lane (ASYNCcnt path, CDNA5)
__device__ __forceinline__ void async_g2l_b128(unsigned ldsoff, const void* gaddr) {
    asm volatile("global_load_async_to_lds_b128 %0, %1, off"
                 :: "v"(ldsoff), "v"((unsigned long long)(uintptr_t)gaddr)
                 : "memory");
}
__device__ __forceinline__ void wait_async0() {
    asm volatile("s_wait_asynccnt 0" ::: "memory");
}

// ---- K1: convert & transpose weights to bf16 -----------------------------
// W1t[n*256 + k] = bf16(W1[k*256 + n])      (65536 elems)
// Wft[n*1024 + k] = bf16(Wf[k*256 + n])     (262144 elems)
__global__ void k_convert(const float* __restrict__ W1, const float* __restrict__ Wf,
                          unsigned short* __restrict__ W1t, unsigned short* __restrict__ Wft) {
    int i = blockIdx.x * 256 + threadIdx.x;
    if (i < 65536) {
        int n = i >> 8, k = i & 255;
        W1t[i] = (unsigned short)f2bf1(W1[k * 256 + n]);
    } else {
        int j = i - 65536;
        if (j < 262144) {
            int n = j >> 10, k = j & 1023;
            Wft[j] = (unsigned short)f2bf1(Wf[k * 256 + n]);
        }
    }
}

// ---- K2: gate = leakyrelu(x@W1+b1)@W2 + b2 via bf16 WMMA -----------------
// block = 256 threads = 8 waves; each wave handles 16 rows.
// W1t staged through 64KB LDS in two 128-column phases via async DMA.
// Two n-tiles processed per iteration -> two independent WMMA chains so
// ds_load latency overlaps matrix ops.
__global__ void k_gate(const float* __restrict__ x, const unsigned short* __restrict__ W1t,
                       const float* __restrict__ b1, const float* __restrict__ W2,
                       const float* __restrict__ b2, const int* __restrict__ batch,
                       float* __restrict__ gate, unsigned* __restrict__ gmaxk, int N) {
    extern __shared__ unsigned short ldsW[];  // 128 * 256 bf16 = 64 KB
    const unsigned ldsBase = (unsigned)(uintptr_t)ldsW;  // flat[31:0] == LDS offset

    const int w = threadIdx.x >> 5;
    const int lane = threadIdx.x & 31;
    const int hs = lane >> 4;        // half-select (K block)
    const int m = lane & 15;         // row within tile / column within tile
    const int rBase = blockIdx.x * 128 + w * 16;

    float gAcc[8];
#pragma unroll
    for (int v = 0; v < 8; ++v) gAcc[v] = 0.f;

    // issue async DMA of one 128-column slice of W1t (64 KB) into LDS
    auto stage = [&](int phase) {
        const char* src = (const char*)(W1t + (size_t)phase * 128 * 256) + threadIdx.x * 16;
        unsigned dst = ldsBase + threadIdx.x * 16;
#pragma unroll
        for (int i = 0; i < 16; ++i) {
            async_g2l_b128(dst, src);
            src += 4096;
            dst += 4096;
        }
    };

    // --- kick off phase-0 weight DMA; overlap with A-fragment build -------
    stage(0);

    // hoist all 8 A-fragments for this wave's 16 rows (fp32 -> bf16)
    int rowA = rBase + m;
    if (rowA > N - 1) rowA = N - 1;  // clamp (results for OOB rows never written)
    const float* xr = x + (size_t)rowA * F;

    ABfrag a[8];
#pragma unroll
    for (int kk = 0; kk < 8; ++kk) {
        int kb = kk * 32 + hs * 8;
        float4 p0 = *(const float4*)(xr + kb);
        float4 p1 = *(const float4*)(xr + kb + 4);
        float4 p2 = *(const float4*)(xr + kb + 16);
        float4 p3 = *(const float4*)(xr + kb + 20);
        a[kk].u[0] = packbf(p0.x, p0.y); a[kk].u[1] = packbf(p0.z, p0.w);
        a[kk].u[2] = packbf(p1.x, p1.y); a[kk].u[3] = packbf(p1.z, p1.w);
        a[kk].u[4] = packbf(p2.x, p2.y); a[kk].u[5] = packbf(p2.z, p2.w);
        a[kk].u[6] = packbf(p3.x, p3.y); a[kk].u[7] = packbf(p3.z, p3.w);
    }

    auto process = [&](int phase) {
#pragma unroll 1
        for (int nt8 = 0; nt8 < 8; nt8 += 2) {
            int ncol0 = (phase * 8 + nt8) * 16 + m;  // global output columns
            int ncol1 = ncol0 + 16;
            int nloc0 = nt8 * 16 + m;                // columns within LDS slice
            int nloc1 = nloc0 + 16;

            v8f c0, c1;
#pragma unroll
            for (int v = 0; v < 8; ++v) { c0[v] = 0.f; c1[v] = 0.f; }

#pragma unroll
            for (int kk = 0; kk < 8; ++kk) {
                ABfrag bb0, bb1;
                const uint4* bp0 = (const uint4*)(ldsW + (size_t)nloc0 * 256 + kk * 32 + hs * 8);
                const uint4* bp1 = (const uint4*)(ldsW + (size_t)nloc1 * 256 + kk * 32 + hs * 8);
                bb0.q[0] = bp0[0]; bb0.q[1] = bp0[2];
                bb1.q[0] = bp1[0]; bb1.q[1] = bp1[2];
                c0 = __builtin_amdgcn_wmma_f32_16x16x32_bf16(
                        false, a[kk].v, false, bb0.v, (short)0, c0, false, false);
                c1 = __builtin_amdgcn_wmma_f32_16x16x32_bf16(
                        false, a[kk].v, false, bb1.v, (short)0, c1, false, false);
            }
            float b1v0 = b1[ncol0], w2v0 = W2[ncol0];
            float b1v1 = b1[ncol1], w2v1 = W2[ncol1];
#pragma unroll
            for (int v = 0; v < 8; ++v) {
                float h0 = c0[v] + b1v0;
                h0 = (h0 >= 0.f) ? h0 : NEG_SLOPE * h0;
                float h1 = c1[v] + b1v1;
                h1 = (h1 >= 0.f) ? h1 : NEG_SLOPE * h1;
                gAcc[v] += h0 * w2v0 + h1 * w2v1;
            }
        }
    };

    wait_async0();
    __syncthreads();
    process(0);

    __syncthreads();   // all waves done reading before LDS is overwritten
    stage(1);
    wait_async0();
    __syncthreads();
    process(1);

    // reduce across the 16 columns held by each half-wave
#pragma unroll
    for (int v = 0; v < 8; ++v) {
        float s = gAcc[v];
        s += __shfl_xor(s, 1, 32);
        s += __shfl_xor(s, 2, 32);
        s += __shfl_xor(s, 4, 32);
        s += __shfl_xor(s, 8, 32);
        gAcc[v] = s;
    }
    if (m == 0) {
        float b2v = b2[0];
#pragma unroll
        for (int v = 0; v < 8; ++v) {
            int row = rBase + hs * 8 + v;   // C layout: M = v + 8*(lane>>4)
            if (row < N) {
                float g = gAcc[v] + b2v;
                gate[row] = g;
                atomicMax(&gmaxk[batch[row]], fkey(g));
            }
        }
    }
}

// ---- K3: e = exp(gate - gmax[batch]) -------------------------------------
__global__ void k_exp(const float* __restrict__ gate, const int* __restrict__ batch,
                      const unsigned* __restrict__ gmaxk, float* __restrict__ e, int N) {
    int i = blockIdx.x * 256 + threadIdx.x;
    if (i < N) {
        float gm = fdekey(gmaxk[batch[i]]);
        e[i] = expf(gate[i] - gm);
    }
}

// ---- K4: fused segment sum/max/count + attention numerator/denominator ---
// block = 256 threads (one per column), 64 consecutive rows; batch sorted so
// accumulate per run in registers, flush atomics only on segment boundaries.
__global__ void k_poolatt(const float* __restrict__ x, const int* __restrict__ batch,
                          const float* __restrict__ e,
                          float* __restrict__ seg_sum, unsigned* __restrict__ segmaxk,
                          float* __restrict__ att_num, float* __restrict__ cnt,
                          float* __restrict__ denom, int N) {
    const int col = threadIdx.x;
    const int r0 = blockIdx.x * 64;
    int cur = -1;
    float aSum = 0.f, aAtt = 0.f, aMax = -3.4e38f, aCnt = 0.f, aDen = 0.f;

    for (int r = 0; r < 64; ++r) {
        int row = r0 + r;
        if (row >= N) break;
        int b = batch[row];
        if (b != cur) {
            if (cur >= 0) {
                size_t gi = (size_t)cur * F + col;
                atomicAdd(&seg_sum[gi], aSum);
                atomicAdd(&att_num[gi], aAtt);
                atomicMax(&segmaxk[gi], fkey(aMax));
                if (col == 0) { atomicAdd(&cnt[cur], aCnt); atomicAdd(&denom[cur], aDen); }
            }
            cur = b; aSum = 0.f; aAtt = 0.f; aMax = -3.4e38f; aCnt = 0.f; aDen = 0.f;
        }
        float ev = e[row];
        float v = x[(size_t)row * F + col];
        aSum += v;
        aAtt += ev * v;
        aMax = fmaxf(aMax, v);
        aCnt += 1.f;
        aDen += ev;
    }
    if (cur >= 0) {
        size_t gi = (size_t)cur * F + col;
        atomicAdd(&seg_sum[gi], aSum);
        atomicAdd(&att_num[gi], aAtt);
        atomicMax(&segmaxk[gi], fkey(aMax));
        if (col == 0) { atomicAdd(&cnt[cur], aCnt); atomicAdd(&denom[cur], aDen); }
    }
}

// ---- K5a: assemble cat = [max | mean | sum | att] as bf16 [G, 1024] ------
__global__ void k_cat(const float* __restrict__ seg_sum, const unsigned* __restrict__ segmaxk,
                      const float* __restrict__ att_num, const float* __restrict__ cnt,
                      const float* __restrict__ denom, unsigned short* __restrict__ cat) {
    size_t idx = (size_t)blockIdx.x * 256 + threadIdx.x;  // < G*1024
    int g = (int)(idx >> 10);
    int j = (int)(idx & 1023);
    int sect = j >> 8;
    int col = j & 255;
    float c = cnt[g];
    size_t gi = (size_t)g * F + col;
    float val;
    if (sect == 0)      val = (c > 0.f) ? fdekey(segmaxk[gi]) : 0.f;
    else if (sect == 1) val = seg_sum[gi] / fmaxf(c, 1.f);
    else if (sect == 2) val = seg_sum[gi];
    else                val = att_num[gi] / fmaxf(denom[g], 1e-16f);
    cat[idx] = (unsigned short)f2bf1(val);
}

// ---- K5b: out = cat @ Wf + bf via bf16 WMMA ------------------------------
// grid = (G/128, 256/32). block = 256 thr = 8 waves; wave: 16 rows x 32 cols.
__global__ void k_final(const unsigned short* __restrict__ cat,
                        const unsigned short* __restrict__ Wft,
                        const float* __restrict__ bfb, float* __restrict__ out) {
    extern __shared__ unsigned short ldsWf[];  // 32 x 1024 bf16 = 64 KB
    const unsigned ldsBase = (unsigned)(uintptr_t)ldsWf;
    const int cBase = blockIdx.y * 32;
    {   // async stage of Wft rows [cBase, cBase+32)  (contiguous: n-major layout)
        const char* src = (const char*)(Wft + (size_t)cBase * 1024) + threadIdx.x * 16;
        unsigned dst = ldsBase + threadIdx.x * 16;
#pragma unroll
        for (int i = 0; i < 16; ++i) {
            async_g2l_b128(dst, src);
            src += 4096;
            dst += 4096;
        }
    }
    wait_async0();
    __syncthreads();

    const int w = threadIdx.x >> 5;
    const int lane = threadIdx.x & 31;
    const int hs = lane >> 4;
    const int m = lane & 15;
    const int rBase = blockIdx.x * 128 + w * 16;
    const unsigned short* ar = cat + (size_t)(rBase + m) * 1024;

    v8f c0, c1;
#pragma unroll
    for (int v = 0; v < 8; ++v) { c0[v] = 0.f; c1[v] = 0.f; }

    for (int kk = 0; kk < 32; ++kk) {
        ABfrag a;
        int kb = kk * 32 + hs * 8;
        a.q[0] = *(const uint4*)(ar + kb);        // K kb .. kb+7
        a.q[1] = *(const uint4*)(ar + kb + 16);   // K kb+16 .. kb+23

        int kbb = kk * 32 + hs * 8;
        ABfrag b0, b1;
        const uint4* bp0 = (const uint4*)(ldsWf + (size_t)m * 1024 + kbb);
        b0.q[0] = bp0[0]; b0.q[1] = bp0[2];
        const uint4* bp1 = (const uint4*)(ldsWf + (size_t)(16 + m) * 1024 + kbb);
        b1.q[0] = bp1[0]; b1.q[1] = bp1[2];

        c0 = __builtin_amdgcn_wmma_f32_16x16x32_bf16(
                false, a.v, false, b0.v, (short)0, c0, false, false);
        c1 = __builtin_amdgcn_wmma_f32_16x16x32_bf16(
                false, a.v, false, b1.v, (short)0, c1, false, false);
    }

    float bias0 = bfb[cBase + m];
    float bias1 = bfb[cBase + 16 + m];
#pragma unroll
    for (int v = 0; v < 8; ++v) {
        int rr = rBase + hs * 8 + v;  // M = v + 8*(lane>>4)
        out[(size_t)rr * 256 + cBase + m]      = c0[v] + bias0;
        out[(size_t)rr * 256 + cBase + 16 + m] = c1[v] + bias1;
    }
}

// ---- host launch ---------------------------------------------------------
extern "C" void kernel_launch(void* const* d_in, const int* in_sizes, int n_in,
                              void* d_out, int out_size, void* d_ws, size_t ws_size,
                              hipStream_t stream) {
    const float* x   = (const float*)d_in[0];
    /* d_in[1] = pos is unused by the reference */
    const int*   bat = (const int*)d_in[2];
    const float* W1  = (const float*)d_in[3];
    const float* b1  = (const float*)d_in[4];
    const float* W2  = (const float*)d_in[5];
    const float* b2  = (const float*)d_in[6];
    const float* Wf  = (const float*)d_in[7];
    const float* bfb = (const float*)d_in[8];
    float* out = (float*)d_out;
    const int N = in_sizes[0] / F;

    char* ws = (char*)d_ws;
    size_t off = 0;
    auto alloc = [&](size_t bytes) {
        size_t o = off;
        off += (bytes + 255) & ~(size_t)255;
        return o;
    };
    size_t oW1t  = alloc((size_t)65536 * 2);
    size_t oWft  = alloc((size_t)262144 * 2);
    size_t oGate = alloc((size_t)N * 4);
    size_t oE    = alloc((size_t)N * 4);
    size_t oCat  = alloc((size_t)G * 1024 * 2);
    size_t zeroStart = off;
    size_t oGmax = alloc((size_t)G * 4);
    size_t oCnt  = alloc((size_t)G * 4);
    size_t oDen  = alloc((size_t)G * 4);
    size_t oSum  = alloc((size_t)G * F * 4);
    size_t oMaxk = alloc((size_t)G * F * 4);
    size_t oAtt  = alloc((size_t)G * F * 4);
    size_t zeroBytes = off - zeroStart;

    hipMemsetAsync(ws + zeroStart, 0, zeroBytes, stream);

    k_convert<<<1280, 256, 0, stream>>>(W1, Wf,
        (unsigned short*)(ws + oW1t), (unsigned short*)(ws + oWft));

    k_gate<<<(N + 127) / 128, 256, 65536, stream>>>(
        x, (const unsigned short*)(ws + oW1t), b1, W2, b2, bat,
        (float*)(ws + oGate), (unsigned*)(ws + oGmax), N);

    k_exp<<<(N + 255) / 256, 256, 0, stream>>>(
        (const float*)(ws + oGate), bat, (const unsigned*)(ws + oGmax),
        (float*)(ws + oE), N);

    k_poolatt<<<(N + 63) / 64, 256, 0, stream>>>(
        x, bat, (const float*)(ws + oE),
        (float*)(ws + oSum), (unsigned*)(ws + oMaxk), (float*)(ws + oAtt),
        (float*)(ws + oCnt), (float*)(ws + oDen), N);

    k_cat<<<(G * 1024) / 256, 256, 0, stream>>>(
        (const float*)(ws + oSum), (const unsigned*)(ws + oMaxk),
        (const float*)(ws + oAtt), (const float*)(ws + oCnt),
        (const float*)(ws + oDen), (unsigned short*)(ws + oCat));

    dim3 fg(G / 128, 256 / 32);
    k_final<<<fg, 256, 65536, stream>>>(
        (const unsigned short*)(ws + oCat), (const unsigned short*)(ws + oWft),
        bfb, out);
}
